// LSTMWithLinear_12137577578694
// MI455X (gfx1250) — compile-verified, matrix operand order checked by reading, not run on hardware
//
#include <hip/hip_runtime.h>

// ---------------------------------------------------------------------------
// Problem constants (from reference): B=1024, T=15, I=256, H=512, F=8, G=4H
// ---------------------------------------------------------------------------
constexpr int Bsz = 1024;
constexpr int Tt  = 15;
constexpr int In  = 256;
constexpr int Hd  = 512;
constexpr int Fh  = 8;
constexpr int Gw  = 4 * Hd;   // 2048, PyTorch gate order i,f,g,o
constexpr int MT  = 64;       // batch rows per workgroup in the LSTM kernel

typedef __bf16 v16bf __attribute__((ext_vector_type(16)));
typedef float  v8f   __attribute__((ext_vector_type(8)));
typedef int    v4i_ls __attribute__((vector_size(16)));   // async-LDS builtin arg

union FragAB { v16bf v; uint4 q[2]; unsigned short u[16]; };
union FragC  { v8f v; float f[8]; };

__device__ __forceinline__ unsigned short f2bf(float x) {
  unsigned int u = __float_as_uint(x);
  u += 0x7FFFu + ((u >> 16) & 1u);          // round-to-nearest-even
  return (unsigned short)(u >> 16);
}
__device__ __forceinline__ float bf2f(unsigned short h) {
  return __uint_as_float(((unsigned int)h) << 16);
}

// CDNA5 has a hardware V_TANH_F32 (TRANS op, co-executes with WMMA).
__device__ __forceinline__ float fast_tanhf(float x) {
#if __has_builtin(__builtin_amdgcn_tanhf)
  return __builtin_amdgcn_tanhf(x);
#elif __has_builtin(__builtin_amdgcn_tanh_f32)
  return __builtin_amdgcn_tanh_f32(x);
#else
  const float e = __expf(2.0f * x);
  return 1.0f - 2.0f / (e + 1.0f);
#endif
}
__device__ __forceinline__ float sigmoidf(float x) {
  return 0.5f * fast_tanhf(0.5f * x) + 0.5f;   // sigmoid via hw tanh
}

#if __has_builtin(__builtin_amdgcn_global_load_async_to_lds_b128) && \
    __has_builtin(__builtin_amdgcn_s_wait_asynccnt)
#define HAVE_ASYNC_LDS 1
#else
#define HAVE_ASYNC_LDS 0
#endif

// ---------------------------------------------------------------------------
// fp32 -> bf16 conversion (weights / X staged once per launch)
// ---------------------------------------------------------------------------
__global__ void cvt_f32_bf16(const float* __restrict__ src,
                             unsigned short* __restrict__ dst, int n) {
  int i = blockIdx.x * blockDim.x + threadIdx.x;
  int stride = gridDim.x * blockDim.x;
  for (; i < n; i += stride) dst[i] = f2bf(src[i]);
}

// ---------------------------------------------------------------------------
// One LSTM cell step for a MT(=64) x Hd slab.
// gates = [x_t | h] @ [Wih | Whh]^T + b ; all GEMM work on bf16 WMMA.
// Wave w owns the disjoint 64-wide H strip [w*64, w*64+64) over ALL 4
// 16-row M tiles: every weight B-fragment is loaded exactly once per
// workgroup and feeds 4 WMMAs (dup=1). Per K-step all 4 gate B-fragments
// are loaded first (one global wait per 16 WMMAs); then each m-tile's
// A-fragment feeds 4 back-to-back WMMAs (one LDS wait per 4 WMMAs).
// Cell state lives in LDS f32; the asm memory clobber between nti tiles
// stops cross-tile load CSE (which previously caused scratch spills).
// ---------------------------------------------------------------------------
template <int KIN>
__device__ __forceinline__ void lstm_step(
    const unsigned short* __restrict__ xsrc, unsigned xstride,
    const unsigned short* __restrict__ hcur,           // LDS MT x Hd (bf16)
    unsigned short* __restrict__ hnext,                // LDS MT x Hd (bf16)
    float* __restrict__ cls,                           // LDS MT x Hd (f32)
    const unsigned short* __restrict__ Wih,            // [Gw x KIN] bf16 (L2)
    const unsigned short* __restrict__ Whh,            // [Gw x Hd]  bf16 (L2)
    const float* __restrict__ bias,                    // [Gw]
    unsigned short* __restrict__ yg, unsigned ystride) // global bf16 out
{
  const unsigned tid  = threadIdx.x;
  const unsigned lane = tid & 31u;
  const unsigned wave = tid >> 5;
  const unsigned lm   = lane & 15u;
  const unsigned hiw  = (lane >> 4) * 8u;
  const unsigned nt0  = wave * 64u;            // this wave's H strip

#pragma unroll
  for (unsigned nti = 0; nti < 4; ++nti) {
    // Barrier against cross-tile load CSE / hoisting (register pressure).
    asm volatile("" ::: "memory");
    const unsigned nt = nt0 + nti * 16u;
    FragC acc[4][4];                     // [m-tile][gate]
#pragma unroll
    for (int m = 0; m < 4; ++m)
#pragma unroll
      for (int g = 0; g < 4; ++g)
#pragma unroll
        for (int r = 0; r < 8; ++r) acc[m][g].f[r] = 0.0f;

    // ---- x part of K loop (input projection folded in) ----
#pragma unroll 2
    for (unsigned k0 = 0; k0 < (unsigned)KIN; k0 += 32u) {
      FragAB bfr[4];                     // all 4 gate B-fragments resident
#pragma unroll
      for (unsigned g = 0; g < 4; ++g) {
        const unsigned bo = (g * (unsigned)Hd + nt + lm) * (unsigned)KIN + k0 + hiw;
        bfr[g].q[0] = *(const uint4*)(Wih + bo);
        bfr[g].q[1] = *(const uint4*)(Wih + bo + 16u);
      }
#pragma unroll
      for (unsigned m = 0; m < 4; ++m) {
        FragAB a;
        const unsigned ao = (m * 16u + lm) * xstride + k0 + hiw;
        a.q[0] = *(const uint4*)(xsrc + ao);
        a.q[1] = *(const uint4*)(xsrc + ao + 16u);
#pragma unroll
        for (unsigned g = 0; g < 4; ++g)
          acc[m][g].v = __builtin_amdgcn_wmma_f32_16x16x32_bf16(
              false, a.v, false, bfr[g].v, (short)0, acc[m][g].v, false, false);
      }
    }
    // ---- h part of K loop (recurrent projection) ----
#pragma unroll 2
    for (unsigned k0 = 0; k0 < (unsigned)Hd; k0 += 32u) {
      FragAB bfr[4];
#pragma unroll
      for (unsigned g = 0; g < 4; ++g) {
        const unsigned bo = (g * (unsigned)Hd + nt + lm) * (unsigned)Hd + k0 + hiw;
        bfr[g].q[0] = *(const uint4*)(Whh + bo);
        bfr[g].q[1] = *(const uint4*)(Whh + bo + 16u);
      }
#pragma unroll
      for (unsigned m = 0; m < 4; ++m) {
        FragAB a;
        const unsigned ao = (m * 16u + lm) * (unsigned)Hd + k0 + hiw;
        a.q[0] = *(const uint4*)(hcur + ao);
        a.q[1] = *(const uint4*)(hcur + ao + 16u);
#pragma unroll
        for (unsigned g = 0; g < 4; ++g)
          acc[m][g].v = __builtin_amdgcn_wmma_f32_16x16x32_bf16(
              false, a.v, false, bfr[g].v, (short)0, acc[m][g].v, false, false);
      }
    }

    // ---- gate nonlinearities + cell/hidden update (lane-local cols) ----
    const unsigned col = nt + lm;
    const float bI = bias[0 * Hd + col];
    const float bF = bias[1 * Hd + col];
    const float bG = bias[2 * Hd + col];
    const float bO = bias[3 * Hd + col];
#pragma unroll
    for (unsigned m = 0; m < 4; ++m) {
#pragma unroll
      for (unsigned r = 0; r < 8; ++r) {
        const unsigned row = m * 16u + r + hiw;   // C/D layout row
        const float iv = sigmoidf(acc[m][0].f[r] + bI);
        const float fv = sigmoidf(acc[m][1].f[r] + bF);
        const float gv = fast_tanhf(acc[m][2].f[r] + bG);
        const float ov = sigmoidf(acc[m][3].f[r] + bO);
        const float cn = fv * cls[row * (unsigned)Hd + col] + iv * gv;
        const float hn = ov * fast_tanhf(cn);
        cls[row * (unsigned)Hd + col] = cn;
        const unsigned short hb = f2bf(hn);
        hnext[row * (unsigned)Hd + col] = hb;
        yg[row * ystride + col] = hb;
      }
    }
  }
}

// ---------------------------------------------------------------------------
// One full LSTM layer pass (all T steps) for a (head, 64-batch-row) slab.
// grid = F * (B/MT). LDS: h double buffer (128 KB) + c f32 (128 KB) =
// 256 KB (needs CDNA5's 320 KB WGP LDS). Layer input rows come from global
// (X for layer 0 with xFStride=0, Y0 for layer 1); every step's hidden
// output is written to Yout[f,b,t,:]. No inter-workgroup sync needed.
// ---------------------------------------------------------------------------
template <int KIN>
__global__ __launch_bounds__(256) void lstm_layer(
    const unsigned short* __restrict__ Xin,   // layer input rows (bf16)
    size_t xFStride,                          // 0 or B*T*KIN (per-head stride)
    const unsigned short* __restrict__ Wih,   // [F][Gw][KIN]
    const unsigned short* __restrict__ Whh,   // [F][Gw][Hd]
    const float* __restrict__ bias,           // [F][Gw]
    unsigned short* __restrict__ Yout)        // [F][B][T][Hd] bf16
{
  extern __shared__ char smem[];
  float* cls = (float*)smem;                        // MT*Hd f32
  unsigned short* ha = (unsigned short*)(cls + MT * Hd);
  unsigned short* hb = ha + MT * Hd;

  const int f   = blockIdx.x / (Bsz / MT);
  const int mb  = blockIdx.x % (Bsz / MT);
  const int b0r = mb * MT;
  const int tid = threadIdx.x;

  for (int i = tid; i < MT * Hd; i += blockDim.x) { cls[i] = 0.f; ha[i] = 0; }
  __syncthreads();

  const unsigned short* wih = Wih + (size_t)f * Gw * KIN;
  const unsigned short* whh = Whh + (size_t)f * Gw * Hd;
  const float* bs = bias + (size_t)f * Gw;

  unsigned short *hc = ha, *hn = hb;
  for (int t = 0; t < Tt; ++t) {
    const unsigned short* xg =
        Xin + (size_t)f * xFStride + ((size_t)b0r * Tt + t) * KIN;
    unsigned short* yg = Yout + (((size_t)f * Bsz + b0r) * Tt + t) * Hd;
    lstm_step<KIN>(xg, (unsigned)(Tt * KIN), hc, hn, cls, wih, whh, bs,
                   yg, (unsigned)(Tt * Hd));
    __syncthreads();
    unsigned short* tmp = hc; hc = hn; hn = tmp;
  }
}

// ---------------------------------------------------------------------------
// Head GEMM: H1[f,b,t,o] = Y[f,b,t,:] @ A1[o,:] + c1[o]  (bf16 WMMA),
// plus per-(f,t) sum / sum-of-squares for the BatchNorm (train-mode stats
// over (b,o)). The 64x512 Y slab is staged once into LDS (async-to-LDS on
// CDNA5); each wave owns a disjoint 64-wide o-strip over all 4 M tiles.
// Per K-step: load B + all 4 A-fragments, then 4 back-to-back WMMAs.
// Stats via wave shfl-reduce + global f32 atomics.
// ---------------------------------------------------------------------------
__global__ __launch_bounds__(256) void head_gemm_stats(
    const unsigned short* __restrict__ Y,    // [F][B][T][Hd]
    const unsigned short* __restrict__ A1b,  // [Hd][Hd] bf16
    const float* __restrict__ c1,            // [Hd]
    unsigned short* __restrict__ H1,         // [F][B][T][Hd] bf16
    float* __restrict__ Sum, float* __restrict__ Sqs)   // [F*T]
{
  extern __shared__ char smem[];
  unsigned short* yls = (unsigned short*)smem;   // 64 x Hd bf16 = 64 KB

  const int ft  = blockIdx.x;               // f*T + t
  const int f   = ft / Tt, t = ft % Tt;
  const unsigned bb = blockIdx.y * 64;      // 64-row batch tile
  const unsigned tid = threadIdx.x, lane = tid & 31u, wave = tid >> 5;
  const unsigned lm  = lane & 15u, hiw = (lane >> 4) * 8u;

  // stage Y slab into LDS (read from L2/HBM exactly once per block)
#if HAVE_ASYNC_LDS
  for (unsigned idx = tid; idx < 64u * Hd / 8u; idx += blockDim.x) {
    const unsigned r  = idx / (Hd / 8);
    const unsigned cc = idx % (Hd / 8);
    const unsigned go = (((unsigned)f * Bsz + bb + r) * Tt + t) * Hd + cc * 8u;
    __builtin_amdgcn_global_load_async_to_lds_b128(
        (v4i_ls*)(Y + go), (v4i_ls*)(yls + r * Hd + cc * 8u), 0, 0);
  }
  __builtin_amdgcn_s_wait_asynccnt(0);
#else
  for (unsigned idx = tid; idx < 64u * Hd / 8u; idx += blockDim.x) {
    const unsigned r  = idx / (Hd / 8);
    const unsigned cc = idx % (Hd / 8);
    const unsigned go = (((unsigned)f * Bsz + bb + r) * Tt + t) * Hd + cc * 8u;
    *(uint4*)(yls + r * Hd + cc * 8u) = *(const uint4*)(Y + go);
  }
#endif
  __syncthreads();

  float ssum = 0.f, ssq = 0.f;
  const unsigned o0base = wave * 64u;       // this wave's o strip
  for (unsigned ot = 0; ot < 4; ++ot) {
    asm volatile("" ::: "memory");          // block cross-tile load hoisting
    const unsigned o0 = o0base + ot * 16u;
    FragC acc[4];
#pragma unroll
    for (int m = 0; m < 4; ++m)
#pragma unroll
      for (int r = 0; r < 8; ++r) acc[m].f[r] = 0.0f;

    for (unsigned k0 = 0; k0 < (unsigned)Hd; k0 += 32u) {
      FragAB b;
      const unsigned bo = (o0 + lm) * (unsigned)Hd + k0 + hiw;
      b.q[0] = *(const uint4*)(A1b + bo);
      b.q[1] = *(const uint4*)(A1b + bo + 16u);
      FragAB a[4];
#pragma unroll
      for (unsigned m = 0; m < 4; ++m) {
        const unsigned ao = (m * 16u + lm) * (unsigned)Hd + k0 + hiw;
        a[m].q[0] = *(const uint4*)(yls + ao);
        a[m].q[1] = *(const uint4*)(yls + ao + 16u);
      }
#pragma unroll
      for (unsigned m = 0; m < 4; ++m)
        acc[m].v = __builtin_amdgcn_wmma_f32_16x16x32_bf16(
            false, a[m].v, false, b.v, (short)0, acc[m].v, false, false);
    }
    const unsigned oc = o0 + lm;
    const float bias = c1[oc];
#pragma unroll
    for (unsigned m = 0; m < 4; ++m) {
#pragma unroll
      for (unsigned r = 0; r < 8; ++r) {
        const float v = acc[m].f[r] + bias;
        ssum += v; ssq += v * v;
        const unsigned brow = bb + m * 16u + r + hiw;
        H1[(((unsigned)f * Bsz + brow) * Tt + t) * Hd + oc] = f2bf(v);
      }
    }
  }
#pragma unroll
  for (int off = 16; off > 0; off >>= 1) {
    ssum += __shfl_xor(ssum, off, 32);
    ssq  += __shfl_xor(ssq,  off, 32);
  }
  if (lane == 0) { atomicAdd(&Sum[ft], ssum); atomicAdd(&Sqs[ft], ssq); }
}

__global__ void zero_stats(float* Sum, float* Sqs) {
  const int i = threadIdx.x;
  if (i < Fh * Tt) { Sum[i] = 0.f; Sqs[i] = 0.f; }
}

__global__ void finalize_stats(const float* __restrict__ Sum,
                               const float* __restrict__ Sqs,
                               float* __restrict__ Mean,
                               float* __restrict__ Scale) {
  const int i = threadIdx.x;
  if (i < Fh * Tt) {
    const float inv = 1.0f / (float)((size_t)Bsz * Hd);
    const float m   = Sum[i] * inv;
    const float var = Sqs[i] * inv - m * m;
    Mean[i]  = m;
    Scale[i] = rsqrtf(var + 1e-5f);
  }
}

// out[b,t,f] = c2 + A2 . relu( (H1 - mean)*rsqrt(var+eps)*gamma[t] + beta[t] )
__global__ __launch_bounds__(256) void head_out(
    const unsigned short* __restrict__ H1,   // [F][B][T][Hd] bf16
    const float* __restrict__ Mean, const float* __restrict__ Scale,
    const float* __restrict__ gamma, const float* __restrict__ beta,
    const float* __restrict__ A2, const float* __restrict__ c2,
    float* __restrict__ out)                 // [B][T][F] f32
{
  const unsigned tid = threadIdx.x, lane = tid & 31u, wave = tid >> 5;
  const unsigned n = blockIdx.x * 8 + wave;  // linear output index
  if (n >= (unsigned)(Bsz * Tt * Fh)) return;
  const unsigned f = n % Fh;
  const unsigned t = (n / Fh) % Tt;
  const unsigned b = n / (Fh * Tt);
  const float m  = Mean[f * Tt + t], s = Scale[f * Tt + t];
  const float gm = gamma[t], bt = beta[t];
  const unsigned hbase = ((f * Bsz + b) * Tt + t) * Hd;
  float acc = 0.f;
  for (unsigned k = lane; k < (unsigned)Hd; k += 32u) {
    float v = (bf2f(H1[hbase + k]) - m) * s * gm + bt;
    v = v > 0.f ? v : 0.f;
    acc += v * A2[k];
  }
#pragma unroll
  for (int off = 16; off > 0; off >>= 1) acc += __shfl_xor(acc, off, 32);
  if (lane == 0) out[n] = acc + c2[0];
}

// ---------------------------------------------------------------------------
// Host-side launcher
// ---------------------------------------------------------------------------
extern "C" void kernel_launch(void* const* d_in, const int* in_sizes, int n_in,
                              void* d_out, int out_size, void* d_ws, size_t ws_size,
                              hipStream_t stream) {
  (void)in_sizes; (void)n_in; (void)out_size; (void)ws_size;

  const float* X     = (const float*)d_in[0];
  const float* Wih0  = (const float*)d_in[1];
  const float* Whh0  = (const float*)d_in[2];
  const float* b0    = (const float*)d_in[3];
  const float* Wih1  = (const float*)d_in[4];
  const float* Whh1  = (const float*)d_in[5];
  const float* b1    = (const float*)d_in[6];
  const float* A1    = (const float*)d_in[7];
  const float* c1    = (const float*)d_in[8];
  const float* gamma = (const float*)d_in[9];
  const float* beta  = (const float*)d_in[10];
  const float* A2    = (const float*)d_in[11];
  const float* c2    = (const float*)d_in[12];
  float* out = (float*)d_out;

  char* ws = (char*)d_ws;
  size_t off = 0;
  auto alloc = [&](size_t bytes) -> char* {
    char* p = ws + off;
    off += (bytes + 255) & ~(size_t)255;
    return p;
  };

  const size_t nX  = (size_t)Bsz * Tt * In;
  const size_t nW0 = (size_t)Fh * Gw * In;
  const size_t nWH = (size_t)Fh * Gw * Hd;
  const size_t nA1 = (size_t)Hd * Hd;
  const size_t nY  = (size_t)Fh * Bsz * Tt * Hd;

  unsigned short* Xbf   = (unsigned short*)alloc(nX * 2);
  unsigned short* W0b   = (unsigned short*)alloc(nW0 * 2);
  unsigned short* Wh0b  = (unsigned short*)alloc(nWH * 2);
  unsigned short* W1b   = (unsigned short*)alloc(nWH * 2);
  unsigned short* Wh1b  = (unsigned short*)alloc(nWH * 2);
  unsigned short* A1b   = (unsigned short*)alloc(nA1 * 2);
  unsigned short* Y0buf = (unsigned short*)alloc(nY * 2);  // layer-0 hidden
  unsigned short* Ybuf  = (unsigned short*)alloc(nY * 2);  // layer-1 hidden
  unsigned short* H1buf = (unsigned short*)alloc(nY * 2);  // head GEMM out
  float* Sum   = (float*)alloc(Fh * Tt * 4);
  float* Sqs   = (float*)alloc(Fh * Tt * 4);
  float* Mean  = (float*)alloc(Fh * Tt * 4);
  float* Scale = (float*)alloc(Fh * Tt * 4);

  // fp32 -> bf16 staging (X + all GEMM operands)
  auto cvt = [&](const float* s, unsigned short* d, size_t n) {
    int blocks = (int)((n + 255) / 256);
    if (blocks > 2048) blocks = 2048;
    cvt_f32_bf16<<<blocks, 256, 0, stream>>>(s, d, (int)n);
  };
  cvt(X,    Xbf,  nX);
  cvt(Wih0, W0b,  nW0);
  cvt(Whh0, Wh0b, nWH);
  cvt(Wih1, W1b,  nWH);
  cvt(Whh1, Wh1b, nWH);
  cvt(A1,   A1b,  nA1);

  zero_stats<<<1, 128, 0, stream>>>(Sum, Sqs);

  // LSTM layers: 256 KB LDS (c f32 + h double buffer) per workgroup.
  const size_t lds_lstm = (size_t)MT * Hd * 4     // c (f32)
                        + (size_t)2 * MT * Hd * 2; // h double buffer (bf16)
  const int lstm_blocks = Fh * (Bsz / MT);   // 128
  lstm_layer<In><<<lstm_blocks, 256, lds_lstm, stream>>>(
      Xbf, (size_t)0, W0b, Wh0b, b0, Y0buf);
  lstm_layer<Hd><<<lstm_blocks, 256, lds_lstm, stream>>>(
      Y0buf, (size_t)Bsz * Tt * Hd, W1b, Wh1b, b1, Ybuf);

  const size_t lds_head = (size_t)64 * Hd * 2;       // Y slab bf16
  head_gemm_stats<<<dim3(Fh * Tt, Bsz / 64), 256, lds_head, stream>>>(
      Ybuf, A1b, c1, H1buf, Sum, Sqs);

  finalize_stats<<<1, 128, 0, stream>>>(Sum, Sqs, Mean, Scale);

  head_out<<<(Bsz * Tt * Fh) / 8, 256, 0, stream>>>(
      H1buf, Mean, Scale, gamma, beta, A2, c2, out);
}
